// MultiHeadAttention_29257317220694
// MI455X (gfx1250) — matrix operand order repaired and checked by reading, not compiled, not run on hardware
//
#include <hip/hip_runtime.h>
#include <hip/hip_bf16.h>

// ---------------------------------------------------------------------------
// MHA forward for MI455X (gfx1250, wave32)
//   - all GEMMs via v_wmma_f32_16x16x32_bf16 (f32 accumulate)
//   - weight tiles staged to LDS via the Tensor Data Mover (tensor_load_to_lds),
//     double buffered, synchronized with s_wait_tensorcnt + barriers
//   - operand loads grouped ahead of WMMA bursts so loads clause & overlap
// B=4, S=2048, D=1024, H=16, HD=64
// ---------------------------------------------------------------------------

#define Bz   4
#define Sq   2048
#define Dm   1024
#define Hh   16
#define HD   64
#define NTOK (Bz * Sq)            // 8192
#define TOT  (NTOK * Dm)          // 8,388,608 elements

typedef __attribute__((ext_vector_type(16))) __bf16 v16bf;
typedef __attribute__((ext_vector_type(8)))  __bf16 bf16x8;
typedef __attribute__((ext_vector_type(4)))  __bf16 bf16x4;
typedef __attribute__((ext_vector_type(8)))  float  v8f;
typedef __attribute__((ext_vector_type(4)))  unsigned int u32x4;
typedef __attribute__((ext_vector_type(8)))  int   i32x8;
typedef __attribute__((ext_vector_type(4)))  int   i32x4;

#define WMMA_BF16(a, b, c)                                                     \
  __builtin_amdgcn_wmma_f32_16x16x32_bf16(false, (a), false, (b), (short)0,    \
                                          (c), false, false)

static __device__ __forceinline__ v16bf cat8(bf16x8 lo, bf16x8 hi) {
  return __builtin_shufflevector(lo, hi, 0, 1, 2, 3, 4, 5, 6, 7, 8, 9, 10, 11,
                                 12, 13, 14, 15);
}

// ---- Tensor Data Mover: 2-D tile load (bf16 elements) into LDS ------------
// D# per CDNA5 ISA ch.8: group0 = {count, lds_addr, global_addr, type=2},
// group1 = {data_size=1(2B), tensor dims, tile dims, dim0 stride}, groups 2/3
// zero (2-D tensor). Issued by one wave; tracked with TENSORcnt.
static __device__ __forceinline__ void tdm_load_tile_bf16(
    unsigned int lds_off, const void* gptr, unsigned int tile_d0,
    unsigned int tile_d1, unsigned int tensor_d0, unsigned int tensor_d1,
    unsigned int stride0) {
  unsigned long long ga = (unsigned long long)(uintptr_t)gptr;
  u32x4 g0;
  g0[0] = 1u;                                     // count=1, user descriptor
  g0[1] = lds_off;                                // lds_addr (bytes)
  g0[2] = (unsigned int)(ga & 0xffffffffu);       // global_addr[31:0]
  g0[3] = (unsigned int)((ga >> 32) & 0x01ffffffu) | (2u << 30);  // type=2
  i32x8 g1;
  g1[0] = (int)(1u << 16);                        // data_size=1 (2 bytes)
  g1[1] = (int)((tensor_d0 & 0xffffu) << 16);     // tensor_dim0[15:0]
  g1[2] = (int)((tensor_d0 >> 16) | ((tensor_d1 & 0xffffu) << 16));
  g1[3] = (int)((tensor_d1 >> 16) | (tile_d0 << 16));  // tile_dim0
  g1[4] = (int)tile_d1;                           // tile_dim1 (tile_dim2=0)
  g1[5] = (int)stride0;                           // tensor_dim0_stride[31:0]
  g1[6] = 0;
  g1[7] = 0;
  i32x4 z4 = {0, 0, 0, 0};
#if defined(__clang_major__) && __clang_major__ >= 23
  i32x8 z8 = {0, 0, 0, 0, 0, 0, 0, 0};
  __builtin_amdgcn_tensor_load_to_lds(g0, g1, z4, z4, z8, 0);
#else
  __builtin_amdgcn_tensor_load_to_lds(g0, g1, z4, z4, 0);
#endif
}

// --------------------------- fp32 -> bf16 conversion -----------------------
__global__ void f32_to_bf16_kernel(const float* __restrict__ src,
                                   __bf16* __restrict__ dst, int n) {
  int i      = (blockIdx.x * blockDim.x + threadIdx.x) * 4;
  int stride = gridDim.x * blockDim.x * 4;
  for (; i + 3 < n; i += stride) {
    float4 v  = *(const float4*)(src + i);
    bf16x4 o  = {(__bf16)v.x, (__bf16)v.y, (__bf16)v.z, (__bf16)v.w};
    *(bf16x4*)(dst + i) = o;
  }
}

// --------------------------- fused QKV projection --------------------------
// Y = x @ W^T (Linear). A rows = tokens (contiguous over D). B column o = W
// row o (contiguous over D), staged per-workgroup into LDS by the TDM.
// Q/K -> [B,H,S,HD], V -> [B,H,HD,S] (transposed for the PV GEMM).
__global__ void qkv_gemm_kernel(const __bf16* __restrict__ xb,
                                const __bf16* __restrict__ wq,
                                const __bf16* __restrict__ wk,
                                const __bf16* __restrict__ wv,
                                __bf16* __restrict__ qo,
                                __bf16* __restrict__ ko,
                                __bf16* __restrict__ vTo) {
  const int lane  = threadIdx.x & 31;
  const int wave  = threadIdx.x >> 5;
  const int half  = lane >> 4;
  const int nIdx  = lane & 15;
  const int mBase = blockIdx.x * 64 + wave * 16;  // token tile base
  const int oBase = blockIdx.y * 64;              // output-feature strip base
  const int which = blockIdx.z;                   // 0=Q 1=K 2=V
  const __bf16* w = (which == 0) ? wq : (which == 1) ? wk : wv;

  // double-buffered weight tile: 64 feature rows x 32 k (bf16) = 4KB each
  __shared__ __align__(128) __bf16 wtile[2][64 * 32];

  const __bf16* arow = xb + (size_t)(mBase + nIdx) * Dm;

  const int nIter = Dm / 32;  // 32 k-chunks
  if (wave == 0) {
    tdm_load_tile_bf16((unsigned int)(uintptr_t)&wtile[0][0],
                       w + (size_t)oBase * Dm, /*tile*/ 32, 64,
                       /*tensor*/ Dm, Dm, /*stride0*/ Dm);
  }

  v8f acc[4] = {};
  for (int i = 0; i < nIter; ++i) {
    const int kk  = i * 32;
    const int cur = i & 1;
    __syncthreads();  // readers of buf[(i+1)&1] (iter i-1) are done
    if (wave == 0) {
      if (i + 1 < nIter) {
        tdm_load_tile_bf16((unsigned int)(uintptr_t)&wtile[cur ^ 1][0],
                           w + (size_t)oBase * Dm + (kk + 32), 32, 64, Dm, Dm,
                           Dm);
        __builtin_amdgcn_s_wait_tensorcnt(1);  // tile i has landed
      } else {
        __builtin_amdgcn_s_wait_tensorcnt(0);
      }
    }
    __syncthreads();  // tile i visible to all waves

    // group ALL operand loads first so they clause & overlap, then burst WMMAs
    v16bf a = cat8(*(const bf16x8*)(arow + kk + half * 16),
                   *(const bf16x8*)(arow + kk + half * 16 + 8));
    v16bf bv[4];
#pragma unroll
    for (int t = 0; t < 4; ++t) {
      const __bf16* brow = &wtile[cur][(t * 16 + nIdx) * 32 + half * 16];
      bv[t] = cat8(*(const bf16x8*)brow, *(const bf16x8*)(brow + 8));
    }
    acc[0] = WMMA_BF16(a, bv[0], acc[0]);
    acc[1] = WMMA_BF16(a, bv[1], acc[1]);
    acc[2] = WMMA_BF16(a, bv[2], acc[2]);
    acc[3] = WMMA_BF16(a, bv[3], acc[3]);
  }

  // C/D map: VGPR j -> row M = j + 8*half, column N = nIdx
#pragma unroll
  for (int t = 0; t < 4; ++t) {
    const int o  = oBase + t * 16 + nIdx;
    const int h  = o >> 6;
    const int hd = o & 63;
#pragma unroll
    for (int j = 0; j < 8; ++j) {
      const int tg  = mBase + j + 8 * half;
      const int b   = tg >> 11;
      const int s   = tg & 2047;
      __bf16    val = (__bf16)acc[t][j];
      if (which == 2) {
        vTo[((size_t)(b * Hh + h) * HD + hd) * Sq + s] = val;   // [B,H,HD,S]
      } else {
        __bf16* dst = (which == 0) ? qo : ko;
        dst[((size_t)(b * Hh + h) * Sq + s) * HD + hd] = val;   // [B,H,S,HD]
      }
    }
  }
}

// --------------------------- flash attention (causal) ----------------------
__global__ void attn_kernel(const __bf16* __restrict__ qb,
                            const __bf16* __restrict__ kb_,
                            const __bf16* __restrict__ vT,
                            __bf16* __restrict__ ctx_out) {
  const int lane = threadIdx.x & 31;
  const int wave = threadIdx.x >> 5;
  const int half = lane >> 4;
  const int nIdx = lane & 15;

  const int wg    = blockIdx.x * 4 + wave;
  const int qTile = wg & 127;               // S/16 = 128
  const int bh    = wg >> 7;                // b*H + h
  const int qBase = qTile * 16;

  __shared__ __bf16 p_lds[4][16][32];       // per-wave P tile (C->A transpose)

  const __bf16* qrow = qb + ((size_t)bh * Sq + qBase + nIdx) * HD;
  const v16bf qa0 = cat8(*(const bf16x8*)(qrow + half * 16),
                         *(const bf16x8*)(qrow + half * 16 + 8));
  const v16bf qa1 = cat8(*(const bf16x8*)(qrow + 32 + half * 16),
                         *(const bf16x8*)(qrow + 32 + half * 16 + 8));

  v8f   ctx0 = {}, ctx1 = {}, ctx2 = {}, ctx3 = {};
  float m[8], l[8];
#pragma unroll
  for (int j = 0; j < 8; ++j) {
    m[j] = -__builtin_inff();
    l[j] = 0.0f;
  }

  const float scale = 0.125f;               // 1/sqrt(HD)
  const int   nkb   = (qBase + 47) >> 5;

  for (int kbi = 0; kbi < nkb; ++kbi) {
    const int kb = kbi * 32;

    if (kbi + 1 < nkb) {
      __builtin_prefetch(kb_ + ((size_t)bh * Sq + kb + 32 + nIdx) * HD, 0, 1);
      __builtin_prefetch(kb_ + ((size_t)bh * Sq + kb + 48 + nIdx) * HD, 0, 1);
    }

    // ---- scores = Q @ K^T: load all four K operands, then 4 WMMAs
    v8f s0 = {}, s1 = {};
    {
      const __bf16* k0 = kb_ + ((size_t)bh * Sq + kb + nIdx) * HD;
      const __bf16* k1 = kb_ + ((size_t)bh * Sq + kb + 16 + nIdx) * HD;
      v16bf b00 = cat8(*(const bf16x8*)(k0 + half * 16),
                       *(const bf16x8*)(k0 + half * 16 + 8));
      v16bf b01 = cat8(*(const bf16x8*)(k0 + 32 + half * 16),
                       *(const bf16x8*)(k0 + 32 + half * 16 + 8));
      v16bf b10 = cat8(*(const bf16x8*)(k1 + half * 16),
                       *(const bf16x8*)(k1 + half * 16 + 8));
      v16bf b11 = cat8(*(const bf16x8*)(k1 + 32 + half * 16),
                       *(const bf16x8*)(k1 + 32 + half * 16 + 8));
      s0 = WMMA_BF16(qa0, b00, s0);
      s1 = WMMA_BF16(qa0, b10, s1);
      s0 = WMMA_BF16(qa1, b01, s0);
      s1 = WMMA_BF16(qa1, b11, s1);
    }

    const bool needMask = (kb + 31 > qBase);
#pragma unroll
    for (int j = 0; j < 8; ++j) {
      float a0 = s0[j] * scale;
      float a1 = s1[j] * scale;
      if (needMask) {
        const int row = qBase + j + 8 * half;
        if (kb + nIdx > row)      a0 = -__builtin_inff();
        if (kb + 16 + nIdx > row) a1 = -__builtin_inff();
      }
      s0[j] = a0;
      s1[j] = a1;
    }

    float alpha[8];
#pragma unroll
    for (int j = 0; j < 8; ++j) {
      float mx = fmaxf(s0[j], s1[j]);
#pragma unroll
      for (int off = 1; off < 16; off <<= 1)
        mx = fmaxf(mx, __shfl_xor(mx, off, 32));
      const float mNew = fmaxf(m[j], mx);
      alpha[j]         = __expf(m[j] - mNew);
      const float p0   = __expf(s0[j] - mNew);
      const float p1   = __expf(s1[j] - mNew);
      float rs         = p0 + p1;
#pragma unroll
      for (int off = 1; off < 16; off <<= 1)
        rs += __shfl_xor(rs, off, 32);
      l[j] = l[j] * alpha[j] + rs;
      m[j] = mNew;
      const int row               = j + 8 * half;
      p_lds[wave][row][nIdx]      = (__bf16)p0;
      p_lds[wave][row][nIdx + 16] = (__bf16)p1;
    }
    asm volatile("s_wait_dscnt 0" ::: "memory");

    const v16bf pa = cat8(*(const bf16x8*)&p_lds[wave][nIdx][half * 16],
                          *(const bf16x8*)&p_lds[wave][nIdx][half * 16 + 8]);

#pragma unroll
    for (int j = 0; j < 8; ++j) {
      ctx0[j] *= alpha[j];
      ctx1[j] *= alpha[j];
      ctx2[j] *= alpha[j];
      ctx3[j] *= alpha[j];
    }

    // ---- ctx += P @ V: load all four V operands, then 4 WMMAs
    const __bf16* vbase = vT + (size_t)bh * HD * Sq + kb + half * 16;
    {
      v16bf vb0 = cat8(*(const bf16x8*)(vbase + (size_t)(0 + nIdx) * Sq),
                       *(const bf16x8*)(vbase + (size_t)(0 + nIdx) * Sq + 8));
      v16bf vb1 = cat8(*(const bf16x8*)(vbase + (size_t)(16 + nIdx) * Sq),
                       *(const bf16x8*)(vbase + (size_t)(16 + nIdx) * Sq + 8));
      v16bf vb2 = cat8(*(const bf16x8*)(vbase + (size_t)(32 + nIdx) * Sq),
                       *(const bf16x8*)(vbase + (size_t)(32 + nIdx) * Sq + 8));
      v16bf vb3 = cat8(*(const bf16x8*)(vbase + (size_t)(48 + nIdx) * Sq),
                       *(const bf16x8*)(vbase + (size_t)(48 + nIdx) * Sq + 8));
      ctx0 = WMMA_BF16(pa, vb0, ctx0);
      ctx1 = WMMA_BF16(pa, vb1, ctx1);
      ctx2 = WMMA_BF16(pa, vb2, ctx2);
      ctx3 = WMMA_BF16(pa, vb3, ctx3);
    }
  }

  float inv[8];
#pragma unroll
  for (int j = 0; j < 8; ++j) inv[j] = 1.0f / l[j];

  const int b = bh >> 4;
  const int h = bh & 15;
#pragma unroll
  for (int t = 0; t < 4; ++t) {
    const v8f& c = (t == 0) ? ctx0 : (t == 1) ? ctx1 : (t == 2) ? ctx2 : ctx3;
#pragma unroll
    for (int j = 0; j < 8; ++j) {
      const int tok = qBase + j + 8 * half;
      ctx_out[((size_t)(b * Sq + tok)) * Dm + h * HD + t * 16 + nIdx] =
          (__bf16)(c[j] * inv[j]);
    }
  }
}

// --------------------------- output projection -----------------------------
__global__ void out_proj_kernel(const __bf16* __restrict__ ctxb,
                                const __bf16* __restrict__ wo,
                                const float* __restrict__ bias,
                                float* __restrict__ out) {
  const int lane  = threadIdx.x & 31;
  const int wave  = threadIdx.x >> 5;
  const int half  = lane >> 4;
  const int nIdx  = lane & 15;
  const int mBase = blockIdx.x * 64 + wave * 16;
  const int oBase = blockIdx.y * 64;

  __shared__ __align__(128) __bf16 wtile[2][64 * 32];

  const __bf16* arow = ctxb + (size_t)(mBase + nIdx) * Dm;

  const int nIter = Dm / 32;
  if (wave == 0) {
    tdm_load_tile_bf16((unsigned int)(uintptr_t)&wtile[0][0],
                       wo + (size_t)oBase * Dm, 32, 64, Dm, Dm, Dm);
  }

  v8f acc[4] = {};
  for (int i = 0; i < nIter; ++i) {
    const int kk  = i * 32;
    const int cur = i & 1;
    __syncthreads();
    if (wave == 0) {
      if (i + 1 < nIter) {
        tdm_load_tile_bf16((unsigned int)(uintptr_t)&wtile[cur ^ 1][0],
                           wo + (size_t)oBase * Dm + (kk + 32), 32, 64, Dm, Dm,
                           Dm);
        __builtin_amdgcn_s_wait_tensorcnt(1);
      } else {
        __builtin_amdgcn_s_wait_tensorcnt(0);
      }
    }
    __syncthreads();

    v16bf a = cat8(*(const bf16x8*)(arow + kk + half * 16),
                   *(const bf16x8*)(arow + kk + half * 16 + 8));
    v16bf bv[4];
#pragma unroll
    for (int t = 0; t < 4; ++t) {
      const __bf16* brow = &wtile[cur][(t * 16 + nIdx) * 32 + half * 16];
      bv[t] = cat8(*(const bf16x8*)brow, *(const bf16x8*)(brow + 8));
    }
    acc[0] = WMMA_BF16(a, bv[0], acc[0]);
    acc[1] = WMMA_BF16(a, bv[1], acc[1]);
    acc[2] = WMMA_BF16(a, bv[2], acc[2]);
    acc[3] = WMMA_BF16(a, bv[3], acc[3]);
  }

#pragma unroll
  for (int t = 0; t < 4; ++t) {
    const int   o  = oBase + t * 16 + nIdx;
    const float bo = bias[o];
#pragma unroll
    for (int j = 0; j < 8; ++j) {
      const int tg             = mBase + j + 8 * half;
      out[(size_t)tg * Dm + o] = acc[t][j] + bo;
    }
  }
}

// --------------------------- host-side launch ------------------------------
extern "C" void kernel_launch(void* const* d_in, const int* in_sizes, int n_in,
                              void* d_out, int out_size, void* d_ws,
                              size_t ws_size, hipStream_t stream) {
  (void)in_sizes; (void)n_in; (void)out_size; (void)ws_size;

  const float* x   = (const float*)d_in[0];
  const float* wq  = (const float*)d_in[1];
  const float* wk  = (const float*)d_in[2];
  const float* wv  = (const float*)d_in[3];
  const float* wo  = (const float*)d_in[4];
  const float* bo  = (const float*)d_in[5];
  float*       out = (float*)d_out;

  __bf16* xb     = (__bf16*)d_ws;
  __bf16* wqb    = xb + (size_t)TOT;
  __bf16* wkb    = wqb + (size_t)Dm * Dm;
  __bf16* wvb    = wkb + (size_t)Dm * Dm;
  __bf16* wob    = wvb + (size_t)Dm * Dm;
  __bf16* qbuf   = wob + (size_t)Dm * Dm;
  __bf16* kbuf   = qbuf + (size_t)TOT;
  __bf16* vTbuf  = kbuf + (size_t)TOT;
  __bf16* ctxbuf = xb;  // alias: xb dead after the QKV GEMM

  f32_to_bf16_kernel<<<4096, 256, 0, stream>>>(x, xb, TOT);
  f32_to_bf16_kernel<<<1024, 256, 0, stream>>>(wq, wqb, Dm * Dm);
  f32_to_bf16_kernel<<<1024, 256, 0, stream>>>(wk, wkb, Dm * Dm);
  f32_to_bf16_kernel<<<1024, 256, 0, stream>>>(wv, wvb, Dm * Dm);
  f32_to_bf16_kernel<<<1024, 256, 0, stream>>>(wo, wob, Dm * Dm);

  dim3 gQKV(NTOK / 64, Dm / 64, 3);
  qkv_gemm_kernel<<<gQKV, 128, 0, stream>>>(xb, wqb, wkb, wvb, qbuf, kbuf,
                                            vTbuf);

  attn_kernel<<<(Bz * Hh * (Sq / 16)) / 4, 128, 0, stream>>>(qbuf, kbuf, vTbuf,
                                                             ctxbuf);

  dim3 gO(NTOK / 64, Dm / 64);
  out_proj_kernel<<<gO, 128, 0, stream>>>(ctxbuf, wob, bo, out);
}
